// AttentionSeqModel_38225208934956
// MI455X (gfx1250) — compile-verified, matrix operand order checked by reading, not run on hardware
//
#include <hip/hip_runtime.h>
#include <hip/hip_bf16.h>

// ---------------------------------------------------------------------------
// AttentionSeqModel on MI455X (gfx1250, wave32, WMMA bf16 16x16x32, f32 acc)
// B=512, L=512, D=128, H=128, A=16.  1024 sequential steps; each step: 32 WGs
// x 128 threads, one WG owns a 16-row batch tile end-to-end.
// ---------------------------------------------------------------------------

#define BATCH 512
#define SEQL  512
#define DOBS  128
#define HID   128
#define H3    384
#define NACT  16

typedef __attribute__((ext_vector_type(16))) __bf16 v16bf;
typedef __attribute__((ext_vector_type(8)))  float  v8f;

static __device__ __forceinline__ __bf16 f2bf(float f) { return (__bf16)f; }

static __device__ __forceinline__ float sigmoidf_(float x) {
    return 1.0f / (1.0f + __expf(-x));
}

// Load a 16x32 bf16 WMMA operand fragment (A or B) from a row-major tile.
// Per CDNA5 ISA 16-bit layout: lane L holds row (rbase + L&15); its 16 elems
// are K = [8*kh, 8*kh+8) and [16+8*kh, 16+8*kh+8) with kh = L>>4, i.e. two
// contiguous 16-byte runs -> two b128 loads.
static __device__ __forceinline__ v16bf frag_ld(const __bf16* base, int rstride,
                                                int rbase, int k0) {
    const int lane = threadIdx.x & 31;
    const int row  = rbase + (lane & 15);
    const int kh   = lane >> 4;
    const __bf16* p = base + (size_t)row * rstride + k0 + 8 * kh;
    union { v16bf v; uint4 u[2]; } f;
    f.u[0] = *reinterpret_cast<const uint4*>(p);
    f.u[1] = *reinterpret_cast<const uint4*>(p + 16);
    return f.v;
}

static __device__ __forceinline__ v8f wmma_bf16(v16bf a, v16bf b, v8f c) {
    // (neg_a, A, neg_b, B, c_mod, C, reuse_a, reuse_b)
    return __builtin_amdgcn_wmma_f32_16x16x32_bf16(false, a, false, b,
                                                   (short)0, c, false, false);
}

// ---------------------------------------------------------------------------
// Prologue kernels
// ---------------------------------------------------------------------------
__global__ void cvt_f32_bf16(const float* __restrict__ s, __bf16* __restrict__ d, int n) {
    int i = blockIdx.x * blockDim.x + threadIdx.x;
    if (i < n) d[i] = f2bf(s[i]);
}

// dec_emb_W is (128 x 16); pad K to 32 with zeros so WMMA K-chunks work.
__global__ void pad_dec_emb(const float* __restrict__ s, __bf16* __restrict__ d) {
    int i = blockIdx.x * blockDim.x + threadIdx.x;
    if (i < 128 * 32) {
        int r = i >> 5, k = i & 31;
        d[i] = (k < 16) ? f2bf(s[r * 16 + k]) : (__bf16)0.0f;
    }
}

__global__ void zero_state(float* __restrict__ h, float* __restrict__ inp) {
    int i = blockIdx.x * blockDim.x + threadIdx.x;
    if (i < BATCH * HID) h[i] = 0.0f;
    if (i < BATCH * NACT) inp[i] = 0.0f;
}

// ---------------------------------------------------------------------------
// Encoder step: emb = x_t @ We^T + be ; h = GRU(emb, h) ; encT[:,t] = h[0]
// grid = 32 WGs (16 rows each) x 128 threads (4 waves)
// ---------------------------------------------------------------------------
__global__ __launch_bounds__(128) void enc_step(
    const float* __restrict__ obs, int t,
    const __bf16* __restrict__ Wemb, const float* __restrict__ bemb,
    const __bf16* __restrict__ Wih,  const __bf16* __restrict__ Whh,
    const float* __restrict__ bih,   const float* __restrict__ bhh,
    float* __restrict__ h, __bf16* __restrict__ encT)
{
    __shared__ alignas(16) char smem[12288 + 32768];
    __bf16* xbf   = (__bf16*)(smem);            // 16x128 bf16
    __bf16* embbf = (__bf16*)(smem + 4096);     // 16x128 bf16
    __bf16* hbf   = (__bf16*)(smem + 8192);     // 16x128 bf16
    float*  S     = (float*)(smem + 12288);     // 4 x (16x128) f32: Sr,Sz,Si,Sh

    const int tid = threadIdx.x, wid = tid >> 5, lane = tid & 31;
    const int gr0 = blockIdx.x * 16;

    // stage x_t and h tiles into LDS as bf16
    for (int i = tid; i < 16 * 128; i += 128) {
        int m = i >> 7, d = i & 127;
        xbf[i] = f2bf(obs[((size_t)(gr0 + m) * SEQL + t) * DOBS + d]);
        hbf[i] = f2bf(h[(gr0 + m) * HID + d]);
    }
    __syncthreads();

    // emb GEMM: M16 x N128 x K128, 2 N-tiles per wave
    {
        const int col = lane & 15, hi = lane >> 4;
        #pragma unroll
        for (int tl = 0; tl < 2; ++tl) {
            int n0 = (wid * 2 + tl) * 16;
            v8f acc = {};
            #pragma unroll
            for (int k0 = 0; k0 < 128; k0 += 32)
                acc = wmma_bf16(frag_ld(xbf, 128, 0, k0),
                                frag_ld(Wemb, 128, n0, k0), acc);
            int n = n0 + col;
            float bb = bemb[n];
            #pragma unroll
            for (int r = 0; r < 8; ++r)
                embbf[(r + 8 * hi) * 128 + n] = f2bf(acc[r] + bb);
        }
    }
    __syncthreads();

    // gi = emb @ Wih^T, gh = h @ Whh^T : N=384, 6 tiles per wave, two passes
    {
        const int col = lane & 15, hi = lane >> 4;
        #pragma unroll
        for (int pass = 0; pass < 2; ++pass) {
            const __bf16* Aact = pass ? hbf : embbf;
            const __bf16* W    = pass ? Whh : Wih;
            const float*  bias = pass ? bhh : bih;
            #pragma unroll
            for (int tl = 0; tl < 6; ++tl) {
                int n0 = (wid * 6 + tl) * 16;
                v8f acc = {};
                #pragma unroll
                for (int k0 = 0; k0 < 128; k0 += 32)
                    acc = wmma_bf16(frag_ld(Aact, 128, 0, k0),
                                    frag_ld(W, 128, n0, k0), acc);
                int n = n0 + col, g = n >> 7, f = n & 127;
                float bb = bias[n];
                #pragma unroll
                for (int r = 0; r < 8; ++r) {
                    int M = r + 8 * hi;
                    float v = acc[r] + bb;
                    if (pass == 0) {
                        if (g < 2) S[g * 2048 + M * 128 + f] = v;       // Sr,Sz
                        else       S[2 * 2048 + M * 128 + f] = v;       // Si
                    } else {
                        if (g < 2) S[g * 2048 + M * 128 + f] += v;
                        else       S[3 * 2048 + M * 128 + f] = v;       // Sh
                    }
                }
            }
        }
    }
    __syncthreads();

    // GRU gates + state update
    for (int i = tid; i < 16 * 128; i += 128) {
        int m = i >> 7, f = i & 127;
        float r  = sigmoidf_(S[m * 128 + f]);
        float z  = sigmoidf_(S[2048 + m * 128 + f]);
        float nn = tanhf(S[4096 + m * 128 + f] + r * S[6144 + m * 128 + f]);
        float hv = h[(gr0 + m) * HID + f];
        float h2 = (1.0f - z) * nn + z * hv;
        h[(gr0 + m) * HID + f] = h2;
        if (blockIdx.x == 0 && m == 0)
            encT[(size_t)f * SEQL + t] = f2bf(h2);   // enc_outs transposed
    }
}

// ---------------------------------------------------------------------------
// Decoder step: emb -> attn(softmax) -> applied -> comb(relu) -> GRU -> out
// ---------------------------------------------------------------------------
__global__ __launch_bounds__(128) void dec_step(
    const __bf16* __restrict__ Wde,  const float* __restrict__ bde,
    const __bf16* __restrict__ Wattn, const float* __restrict__ battn,
    const __bf16* __restrict__ Wcomb, const float* __restrict__ bcomb,
    const __bf16* __restrict__ Wih,  const __bf16* __restrict__ Whh,
    const float* __restrict__ bih,   const float* __restrict__ bhh,
    const __bf16* __restrict__ Wout, const float* __restrict__ bout,
    const __bf16* __restrict__ encT,
    float* __restrict__ h, float* __restrict__ inp,
    float* __restrict__ out_final)
{
    __shared__ alignas(16) char smem[58752];
    __bf16* catA  = (__bf16*)(smem);            // 16x256 [emb | h]
    __bf16* catB  = (__bf16*)(smem + 8192);     // 16x256 [emb | applied]
    __bf16* h2bf  = (__bf16*)(smem + 16384);    // 16x128
    __bf16* obf   = (__bf16*)(smem + 20480);    // 16x128
    __bf16* awbf  = (__bf16*)(smem + 24576);    // 16x512 attn weights (softmaxed)
    float*  S     = (float*)(smem + 24576);     // 4x(16x128) f32 (reuses awbf region later)
    __bf16* inpbf = (__bf16*)(smem + 57344);    // 16x32 (K padded)
    float*  lbuf  = (float*)(smem + 57344);     // 16x16 logits (reuses inpbf later)
    float*  wred  = (float*)(smem + 58368);     // 16x4 cross-wave reduce
    float*  rowred= (float*)(smem + 58624);     // 16

    const int tid = threadIdx.x, wid = tid >> 5, lane = tid & 31;
    const int col = lane & 15, hi = lane >> 4;
    const int gr0 = blockIdx.x * 16;

    // stage inp (zero-padded to K=32) and h into LDS
    for (int i = tid; i < 16 * 32; i += 128) {
        int m = i >> 5, k = i & 31;
        inpbf[i] = (k < NACT) ? f2bf(inp[(gr0 + m) * NACT + k]) : (__bf16)0.0f;
    }
    for (int i = tid; i < 16 * 128; i += 128) {
        int m = i >> 7, f = i & 127;
        catA[m * 256 + 128 + f] = f2bf(h[(gr0 + m) * HID + f]);
    }
    __syncthreads();

    // emb = inp @ Wde^T + bde -> catA[:,0:128]
    #pragma unroll
    for (int tl = 0; tl < 2; ++tl) {
        int n0 = (wid * 2 + tl) * 16;
        v8f acc = {};
        acc = wmma_bf16(frag_ld(inpbf, 32, 0, 0), frag_ld(Wde, 32, n0, 0), acc);
        int n = n0 + col;
        float bb = bde[n];
        #pragma unroll
        for (int r = 0; r < 8; ++r)
            catA[(r + 8 * hi) * 256 + n] = f2bf(acc[r] + bb);
    }
    __syncthreads();

    // attn logits (kept in accumulators): [emb|h](16x256) @ Wattn^T -> 16x512
    {
        v8f a8[8];
        #pragma unroll
        for (int tl = 0; tl < 8; ++tl) {
            int n0 = wid * 128 + tl * 16;
            v8f acc = {};
            #pragma unroll
            for (int k0 = 0; k0 < 256; k0 += 32)
                acc = wmma_bf16(frag_ld(catA, 256, 0, k0),
                                frag_ld(Wattn, 256, n0, k0), acc);
            float bb = battn[n0 + col];
            #pragma unroll
            for (int r = 0; r < 8; ++r) acc[r] += bb;
            a8[tl] = acc;
        }
        // row max: per-lane over 8 tiles, shfl-xor within the 16-lane half
        float pm[8];
        #pragma unroll
        for (int r = 0; r < 8; ++r) {
            float mx = a8[0][r];
            #pragma unroll
            for (int tl = 1; tl < 8; ++tl) mx = fmaxf(mx, a8[tl][r]);
            #pragma unroll
            for (int mk = 1; mk < 16; mk <<= 1) mx = fmaxf(mx, __shfl_xor(mx, mk, 32));
            pm[r] = mx;
        }
        if (col == 0) {
            #pragma unroll
            for (int r = 0; r < 8; ++r) wred[(r + 8 * hi) * 4 + wid] = pm[r];
        }
        __syncthreads();
        if (tid < 16) {
            float mx = wred[tid * 4];
            for (int w = 1; w < 4; ++w) mx = fmaxf(mx, wred[tid * 4 + w]);
            rowred[tid] = mx;
        }
        __syncthreads();
        // exp + row sum
        #pragma unroll
        for (int r = 0; r < 8; ++r) {
            float rmx = rowred[r + 8 * hi];
            float sm = 0.0f;
            #pragma unroll
            for (int tl = 0; tl < 8; ++tl) {
                float e = __expf(a8[tl][r] - rmx);
                a8[tl][r] = e;
                sm += e;
            }
            #pragma unroll
            for (int mk = 1; mk < 16; mk <<= 1) sm += __shfl_xor(sm, mk, 32);
            pm[r] = sm;
        }
        __syncthreads();            // rowred(max) fully consumed
        if (col == 0) {
            #pragma unroll
            for (int r = 0; r < 8; ++r) wred[(r + 8 * hi) * 4 + wid] = pm[r];
        }
        __syncthreads();
        if (tid < 16) {
            float sm = 0.0f;
            for (int w = 0; w < 4; ++w) sm += wred[tid * 4 + w];
            rowred[tid] = sm;
        }
        __syncthreads();
        // normalized attn weights -> awbf (bf16)
        #pragma unroll
        for (int r = 0; r < 8; ++r) {
            int M = r + 8 * hi;
            float inv = 1.0f / rowred[M];
            #pragma unroll
            for (int tl = 0; tl < 8; ++tl)
                awbf[M * 512 + wid * 128 + tl * 16 + col] = f2bf(a8[tl][r] * inv);
        }
    }
    __syncthreads();

    // catB[:,0:128] = emb ; applied = aw @ enc_outs -> catB[:,128:256]
    for (int i = tid; i < 16 * 128; i += 128) {
        int m = i >> 7, f = i & 127;
        catB[m * 256 + f] = catA[m * 256 + f];
    }
    #pragma unroll
    for (int tl = 0; tl < 2; ++tl) {
        int n0 = wid * 32 + tl * 16;
        v8f acc = {};
        #pragma unroll
        for (int k0 = 0; k0 < 512; k0 += 32)
            acc = wmma_bf16(frag_ld(awbf, 512, 0, k0),
                            frag_ld(encT, 512, n0, k0), acc);
        int n = n0 + col;
        #pragma unroll
        for (int r = 0; r < 8; ++r)
            catB[(r + 8 * hi) * 256 + 128 + n] = f2bf(acc[r]);
    }
    __syncthreads();

    // o = relu([emb|applied] @ Wcomb^T + bcomb) -> obf
    #pragma unroll
    for (int tl = 0; tl < 2; ++tl) {
        int n0 = wid * 32 + tl * 16;
        v8f acc = {};
        #pragma unroll
        for (int k0 = 0; k0 < 256; k0 += 32)
            acc = wmma_bf16(frag_ld(catB, 256, 0, k0),
                            frag_ld(Wcomb, 256, n0, k0), acc);
        int n = n0 + col;
        float bb = bcomb[n];
        #pragma unroll
        for (int r = 0; r < 8; ++r)
            obf[(r + 8 * hi) * 128 + n] = f2bf(fmaxf(acc[r] + bb, 0.0f));
    }
    __syncthreads();

    // GRU: gi = o @ Wih^T, gh = h @ Whh^T  (h bf16 lives in catA[:,128:256])
    #pragma unroll
    for (int pass = 0; pass < 2; ++pass) {
        const __bf16* Aact = pass ? (catA + 128) : obf;
        const int astride  = pass ? 256 : 128;
        const __bf16* W    = pass ? Whh : Wih;
        const float*  bias = pass ? bhh : bih;
        #pragma unroll
        for (int tl = 0; tl < 6; ++tl) {
            int n0 = (wid * 6 + tl) * 16;
            v8f acc = {};
            #pragma unroll
            for (int k0 = 0; k0 < 128; k0 += 32)
                acc = wmma_bf16(frag_ld(Aact, astride, 0, k0),
                                frag_ld(W, 128, n0, k0), acc);
            int n = n0 + col, g = n >> 7, f = n & 127;
            float bb = bias[n];
            #pragma unroll
            for (int r = 0; r < 8; ++r) {
                int M = r + 8 * hi;
                float v = acc[r] + bb;
                if (pass == 0) {
                    if (g < 2) S[g * 2048 + M * 128 + f] = v;
                    else       S[2 * 2048 + M * 128 + f] = v;
                } else {
                    if (g < 2) S[g * 2048 + M * 128 + f] += v;
                    else       S[3 * 2048 + M * 128 + f] = v;
                }
            }
        }
    }
    __syncthreads();

    // gates -> h (global) and h2bf (LDS)
    for (int i = tid; i < 16 * 128; i += 128) {
        int m = i >> 7, f = i & 127;
        float r  = sigmoidf_(S[m * 128 + f]);
        float z  = sigmoidf_(S[2048 + m * 128 + f]);
        float nn = tanhf(S[4096 + m * 128 + f] + r * S[6144 + m * 128 + f]);
        float hv = h[(gr0 + m) * HID + f];
        float h2 = (1.0f - z) * nn + z * hv;
        h[(gr0 + m) * HID + f] = h2;
        h2bf[m * 128 + f] = f2bf(h2);
    }
    __syncthreads();

    // out logits: h2 @ Wout^T + bout (N=16, wave 0 only), then log_softmax
    if (wid == 0) {
        v8f acc = {};
        #pragma unroll
        for (int k0 = 0; k0 < 128; k0 += 32)
            acc = wmma_bf16(frag_ld(h2bf, 128, 0, k0),
                            frag_ld(Wout, 128, 0, k0), acc);
        float bb = bout[col];
        #pragma unroll
        for (int r = 0; r < 8; ++r)
            lbuf[(r + 8 * hi) * 16 + col] = acc[r] + bb;
    }
    __syncthreads();

    if (tid < 16) {
        int m = tid;
        float mx = lbuf[m * 16];
        for (int a = 1; a < NACT; ++a) mx = fmaxf(mx, lbuf[m * 16 + a]);
        float sm = 0.0f;
        for (int a = 0; a < NACT; ++a) sm += __expf(lbuf[m * 16 + a] - mx);
        float lse = mx + __logf(sm);
        for (int a = 0; a < NACT; ++a) {
            float v = lbuf[m * 16 + a] - lse;
            inp[(gr0 + m) * NACT + a] = v;           // fed back next step
            if (out_final) out_final[(gr0 + m) * NACT + a] = v;
        }
    }
}

// ---------------------------------------------------------------------------
// Host side
// ---------------------------------------------------------------------------
extern "C" void kernel_launch(void* const* d_in, const int* in_sizes, int n_in,
                              void* d_out, int out_size, void* d_ws, size_t ws_size,
                              hipStream_t stream) {
    const float* obs      = (const float*)d_in[0];
    const float* enc_embW = (const float*)d_in[1];
    const float* enc_embb = (const float*)d_in[2];
    const float* enc_Wih  = (const float*)d_in[3];
    const float* enc_Whh  = (const float*)d_in[4];
    const float* enc_bih  = (const float*)d_in[5];
    const float* enc_bhh  = (const float*)d_in[6];
    const float* dec_embW = (const float*)d_in[7];
    const float* dec_embb = (const float*)d_in[8];
    const float* attn_W   = (const float*)d_in[9];
    const float* attn_b   = (const float*)d_in[10];
    const float* comb_W   = (const float*)d_in[11];
    const float* comb_b   = (const float*)d_in[12];
    const float* dec_Wih  = (const float*)d_in[13];
    const float* dec_Whh  = (const float*)d_in[14];
    const float* dec_bih  = (const float*)d_in[15];
    const float* dec_bhh  = (const float*)d_in[16];
    const float* out_W    = (const float*)d_in[17];
    const float* out_b    = (const float*)d_in[18];

    char* ws = (char*)d_ws;
    size_t o = 0;
    auto take = [&](size_t bytes) { char* p = ws + o; o += (bytes + 255) & ~size_t(255); return p; };

    __bf16* bWembE = (__bf16*)take(HID * DOBS * 2);     // 128x128
    __bf16* bWihE  = (__bf16*)take(H3 * HID * 2);       // 384x128
    __bf16* bWhhE  = (__bf16*)take(H3 * HID * 2);
    __bf16* bWde   = (__bf16*)take(HID * 32 * 2);       // 128x32 (padded)
    __bf16* bWattn = (__bf16*)take(SEQL * 2 * HID * 2); // 512x256
    __bf16* bWcomb = (__bf16*)take(HID * 2 * HID * 2);  // 128x256
    __bf16* bWihD  = (__bf16*)take(H3 * HID * 2);
    __bf16* bWhhD  = (__bf16*)take(H3 * HID * 2);
    __bf16* bWout  = (__bf16*)take(NACT * HID * 2);     // 16x128
    __bf16* encT   = (__bf16*)take(HID * SEQL * 2);     // 128x512 (enc_outs^T)
    float*  hbuf   = (float*)take(BATCH * HID * 4);
    float*  inpbuf = (float*)take(BATCH * NACT * 4);

    auto cvt = [&](const float* s, __bf16* d, int n) {
        cvt_f32_bf16<<<(n + 255) / 256, 256, 0, stream>>>(s, d, n);
    };
    cvt(enc_embW, bWembE, HID * DOBS);
    cvt(enc_Wih,  bWihE,  H3 * HID);
    cvt(enc_Whh,  bWhhE,  H3 * HID);
    cvt(attn_W,   bWattn, SEQL * 2 * HID);
    cvt(comb_W,   bWcomb, HID * 2 * HID);
    cvt(dec_Wih,  bWihD,  H3 * HID);
    cvt(dec_Whh,  bWhhD,  H3 * HID);
    cvt(out_W,    bWout,  NACT * HID);
    pad_dec_emb<<<(128 * 32 + 255) / 256, 256, 0, stream>>>(dec_embW, bWde);
    zero_state<<<(BATCH * HID + 255) / 256, 256, 0, stream>>>(hbuf, inpbuf);

    // encoder: 512 sequential steps
    for (int t = 0; t < SEQL; ++t)
        enc_step<<<BATCH / 16, 128, 0, stream>>>(obs, t, bWembE, enc_embb,
                                                 bWihE, bWhhE, enc_bih, enc_bhh,
                                                 hbuf, encT);

    // decoder: 512 sequential steps; last step also writes d_out
    for (int t = 0; t < SEQL; ++t)
        dec_step<<<BATCH / 16, 128, 0, stream>>>(bWde, dec_embb,
                                                 bWattn, attn_b,
                                                 bWcomb, comb_b,
                                                 bWihD, bWhhD, dec_bih, dec_bhh,
                                                 bWout, out_b, encT,
                                                 hbuf, inpbuf,
                                                 (t == SEQL - 1) ? (float*)d_out : nullptr);
    (void)in_sizes; (void)n_in; (void)out_size; (void)ws_size;
}
